// ChessMultiStageAttention_77266461655295
// MI455X (gfx1250) — matrix operand order repaired and checked by reading, not compiled, not run on hardware
//
#include <hip/hip_runtime.h>
#include <hip/hip_bf16.h>

typedef __attribute__((ext_vector_type(16))) _Float16 v16h;
typedef __attribute__((ext_vector_type(8)))  _Float16 v8h;
typedef __attribute__((ext_vector_type(8)))  float    v8f;

#define DEVINL __device__ __forceinline__

static constexpr int S  = 64;    // tokens (8*8)
static constexpr int E  = 64;    // embed dim
static constexpr int E3 = 192;   // 3*E
static constexpr int NH = 8;     // heads
static constexpr int HD = 8;     // head dim
static constexpr int NBATCH = 4096;

// ---- dynamic LDS layout (bytes) ----
static constexpr int OFF_XRES = 0;        // f32 [64][64]  x+pos residual
static constexpr int OFF_XN   = 16384;    // f16 [64][64]  layernormed x
static constexpr int OFF_QKV  = 24576;    // f16 [64][192] qkv
static constexpr int OFF_VT   = 49152;    // f16 [64][64]  V transposed: vT[h*8+d][t]
static constexpr int OFF_BIAS = 57344;    // f32 [64][64]  chess bias (aliased w/ final out)
static constexpr int OFF_ASCR = 73728;    // f16 [8 waves][16][64] attn row-block scratch
static constexpr int OFF_OUTP = 90112;    // f16 [64][64]  attention output (pre-proj)
static constexpr int LDS_BYTES = 98304;   // 96 KB (< 320 KB/WGP -> 3 WG/WGP)

// Load WMMA A-operand (16x32 f16) from row-major f16 buffer, rows r0..r0+15,
// K-columns kt*32..kt*32+31.  Per-lane K-sequential:
//  lanes 0-15 : row r0+l,    K = 0..15
//  lanes16-31 : row r0+l-16, K = 16..31
DEVINL v16h load_a16(const _Float16* base, int ld, int r0, int kt, int hl, int lr) {
    const _Float16* p = base + (r0 + lr) * ld + kt * 32 + hl * 16;
    v8h lo = *(const v8h*)p;
    v8h hi = *(const v8h*)(p + 8);
    return __builtin_shufflevector(lo, hi, 0,1,2,3,4,5,6,7,8,9,10,11,12,13,14,15);
}

// Load WMMA B-operand (32x16 f16) for y = x * W^T from row-major f16 weight [O][E]:
// B[k][n] = W[ntile*16+n][kt*32+k].  Per-lane K-sequential:
//  lanes 0-15 : n = l,    K = 0..15   -> W row, e = kt*32 + 0..15   (contiguous)
//  lanes16-31 : n = l-16, K = 16..31  -> W row, e = kt*32 + 16..31  (contiguous)
DEVINL v16h load_bw(const _Float16* w, int ldw, int ntile, int kt, int hl, int lr) {
    const _Float16* p = w + (ntile * 16 + lr) * ldw + kt * 32 + hl * 16;
    v8h lo = *(const v8h*)p;
    v8h hi = *(const v8h*)(p + 8);
    return __builtin_shufflevector(lo, hi, 0,1,2,3,4,5,6,7,8,9,10,11,12,13,14,15);
}

DEVINL v8f wmma_f16(v16h a, v16h b, v8f c) {
    return __builtin_amdgcn_wmma_f32_16x16x32_f16(false, a, false, b, (short)0, c, false, false);
}

// ---- prep: convert weights f32 -> f16 into workspace (shared by all blocks) ----
__global__ void __launch_bounds__(256)
prep_weights(const float* __restrict__ wq, const float* __restrict__ wo,
             _Float16* __restrict__ wq16, _Float16* __restrict__ wo16) {
    int i = blockIdx.x * 256 + threadIdx.x;
    if (i < E3 * E) wq16[i] = (_Float16)wq[i];
    if (i < E * E)  wo16[i] = (_Float16)wo[i];
}

// ---- main fused kernel: one workgroup (8 wave32) per batch element ----
__global__ void __launch_bounds__(256)
chess_attn_kernel(const float* __restrict__ x,    const float* __restrict__ pos,
                  const float* __restrict__ gma,  const float* __restrict__ bta,
                  const float* __restrict__ qkvb, const float* __restrict__ outb,
                  const float* __restrict__ cbias,
                  const _Float16* __restrict__ wq16, const _Float16* __restrict__ wo16,
                  float* __restrict__ outg) {
    extern __shared__ char smem[];
    float*    xres  = (float*)(smem + OFF_XRES);
    _Float16* xn    = (_Float16*)(smem + OFF_XN);
    _Float16* qkv   = (_Float16*)(smem + OFF_QKV);
    _Float16* vT    = (_Float16*)(smem + OFF_VT);
    float*    biasF = (float*)(smem + OFF_BIAS);
    float*    finF  = biasF;                       // alias: bias dead after attention
    _Float16* ascr  = (_Float16*)(smem + OFF_ASCR);
    _Float16* outp  = (_Float16*)(smem + OFF_OUTP);

    const int tid  = threadIdx.x;
    const int wave = tid >> 5, lane = tid & 31;
    const int hl   = lane >> 4, lr = lane & 15;    // half-wave, lane-in-half
    const size_t bbase = (size_t)blockIdx.x * (S * E);

    // prefetch the (L2-resident) f16 weights: emits global_prefetch_b8
    if (tid < E3) __builtin_prefetch(wq16 + tid * E, 0, 3);
    if (tid < E)  __builtin_prefetch(wo16 + tid * E, 0, 3);

    // ---- phase 0a: chess bias -> LDS via async DMA (ASYNCcnt, no VGPR round trip) ----
    {
        unsigned gofs = (unsigned)tid * 16u;                              // byte offset
        unsigned lofs = (unsigned)(size_t)(smem + OFF_BIAS) + gofs;       // LDS byte addr
        #pragma unroll
        for (int i = 0; i < 4; ++i) {
            asm volatile("global_load_async_to_lds_b128 %0, %1, %2"
                         :: "v"(lofs + (unsigned)i * 4096u),
                            "v"(gofs + (unsigned)i * 4096u),
                            "s"(cbias)
                         : "memory");
        }
    }

    // ---- phase 0b: x + pos_embed, transpose [c][s] -> [s][c] into LDS ----
    #pragma unroll
    for (int i = 0; i < 4; ++i) {
        int idx = (i * 256 + tid) * 4;             // idx = c*64 + s0, s0 % 4 == 0
        int c = idx >> 6, s0 = idx & 63;
        float4 xv = *(const float4*)(x + bbase + idx);
        float4 pv = *(const float4*)(pos + idx);
        xres[(s0 + 0) * E + c] = xv.x + pv.x;
        xres[(s0 + 1) * E + c] = xv.y + pv.y;
        xres[(s0 + 2) * E + c] = xv.z + pv.z;
        xres[(s0 + 3) * E + c] = xv.w + pv.w;
    }
    asm volatile("s_wait_asynccnt 0" ::: "memory");   // bias DMA complete
    __syncthreads();

    // ---- phase 1: LayerNorm over E (one token per thread, threads 0..63) ----
    if (tid < S) {
        const float* row = xres + tid * E;
        float sum = 0.f, sq = 0.f;
        #pragma unroll 8
        for (int e = 0; e < E; ++e) { float v = row[e]; sum += v; sq += v * v; }
        float mu  = sum * (1.0f / E);
        float var = sq * (1.0f / E) - mu * mu;
        float rs  = rsqrtf(var + 1e-5f);
        #pragma unroll 8
        for (int e = 0; e < E; ++e)
            xn[tid * E + e] = (_Float16)((row[e] - mu) * rs * gma[e] + bta[e]);
    }
    __syncthreads();

    // ---- phase 2: QKV GEMM  [64x64] x [64x192], 48 tiles, 6 per wave ----
    {
        const int m  = wave & 3;
        const int n0 = (wave >> 2) * 6;
        v16h A0 = load_a16(xn, E, m * 16, 0, hl, lr);
        v16h A1 = load_a16(xn, E, m * 16, 1, hl, lr);
        #pragma unroll
        for (int n = n0; n < n0 + 6; ++n) {
            v16h B0 = load_bw(wq16, E, n, 0, hl, lr);
            v16h B1 = load_bw(wq16, E, n, 1, hl, lr);
            v8f acc = {};
            acc = wmma_f16(A0, B0, acc);
            acc = wmma_f16(A1, B1, acc);
            int col = n * 16 + lr;
            float bc = qkvb[col];
            _Float16 hv[8];
            #pragma unroll
            for (int j = 0; j < 8; ++j) hv[j] = (_Float16)(acc[j] + bc);
            #pragma unroll
            for (int j = 0; j < 8; ++j)
                qkv[(m * 16 + hl * 8 + j) * E3 + col] = hv[j];
            if (n >= 8) {                           // uniform: V tiles -> transposed copy
                #pragma unroll
                for (int j = 0; j < 8; ++j)
                    vT[(col - 2 * E) * S + (m * 16 + hl * 8 + j)] = hv[j];
            }
        }
    }
    __syncthreads();

    // ---- phase 3: attention, one head per wave ----
    {
        const int h = wave;
        const float scale = 0.35355339059327373f;  // 1/sqrt(8)
        _Float16* myscr = ascr + wave * (16 * S);

        // B-operand for Q*K^T: B[d][t] = K[t][d], d padded 8->32 with zeros
        v16h Bk[4];
        #pragma unroll
        for (int nt = 0; nt < 4; ++nt) {
            v8h lo = {}, hi = {};
            if (hl == 0) lo = *(const v8h*)(qkv + (nt * 16 + lr) * E3 + E + h * 8);
            Bk[nt] = __builtin_shufflevector(lo, hi, 0,1,2,3,4,5,6,7,8,9,10,11,12,13,14,15);
        }
        // B-operand for attn*V: B[t][d] = V[t][d], n cols 8..15 zero
        v16h Bv[2];
        #pragma unroll
        for (int kt = 0; kt < 2; ++kt) {
            v8h lo = {}, hi = {};
            if (lr < HD) {
                const _Float16* p = vT + (h * 8 + lr) * S + kt * 32 + hl * 16;
                lo = *(const v8h*)p;
                hi = *(const v8h*)(p + 8);
            }
            Bv[kt] = __builtin_shufflevector(lo, hi, 0,1,2,3,4,5,6,7,8,9,10,11,12,13,14,15);
        }

        #pragma unroll
        for (int m = 0; m < 4; ++m) {
            // A-operand: Q rows m*16.., K-dim padded 8->32 with zeros
            v8h qlo = {}, qz = {};
            if (hl == 0) qlo = *(const v8h*)(qkv + (m * 16 + lr) * E3 + h * 8);
            v16h Aq = __builtin_shufflevector(qlo, qz, 0,1,2,3,4,5,6,7,8,9,10,11,12,13,14,15);

            v8f sc[4];
            #pragma unroll
            for (int nt = 0; nt < 4; ++nt) {
                v8f z = {};
                sc[nt] = wmma_f16(Aq, Bk[nt], z);
            }
            // scale + chess bias
            #pragma unroll
            for (int nt = 0; nt < 4; ++nt)
                #pragma unroll
                for (int j = 0; j < 8; ++j) {
                    int row = m * 16 + hl * 8 + j, col = nt * 16 + lr;
                    sc[nt][j] = sc[nt][j] * scale + biasF[row * E + col];
                }
            // softmax: full row of 64 lives in 4 tiles x 16 lanes of one half-wave
            #pragma unroll
            for (int j = 0; j < 8; ++j) {
                float mx = fmaxf(fmaxf(sc[0][j], sc[1][j]), fmaxf(sc[2][j], sc[3][j]));
                mx = fmaxf(mx, __shfl_xor(mx, 1, 32));
                mx = fmaxf(mx, __shfl_xor(mx, 2, 32));
                mx = fmaxf(mx, __shfl_xor(mx, 4, 32));
                mx = fmaxf(mx, __shfl_xor(mx, 8, 32));
                float e0 = __expf(sc[0][j] - mx), e1 = __expf(sc[1][j] - mx);
                float e2 = __expf(sc[2][j] - mx), e3 = __expf(sc[3][j] - mx);
                float sm = e0 + e1 + e2 + e3;
                sm += __shfl_xor(sm, 1, 32);
                sm += __shfl_xor(sm, 2, 32);
                sm += __shfl_xor(sm, 4, 32);
                sm += __shfl_xor(sm, 8, 32);
                float r = __builtin_amdgcn_rcpf(sm);
                int lrow = hl * 8 + j;
                myscr[lrow * S +  0 + lr] = (_Float16)(e0 * r);
                myscr[lrow * S + 16 + lr] = (_Float16)(e1 * r);
                myscr[lrow * S + 32 + lr] = (_Float16)(e2 * r);
                myscr[lrow * S + 48 + lr] = (_Float16)(e3 * r);
            }
            // attn @ V for this 16-row block
            v16h Aa0 = load_a16(myscr, S, 0, 0, hl, lr);
            v16h Aa1 = load_a16(myscr, S, 0, 1, hl, lr);
            v8f o = {};
            o = wmma_f16(Aa0, Bv[0], o);
            o = wmma_f16(Aa1, Bv[1], o);
            if (lr < HD) {
                #pragma unroll
                for (int j = 0; j < 8; ++j) {
                    int row = m * 16 + hl * 8 + j;
                    outp[row * E + h * 8 + lr] = (_Float16)o[j];
                }
            }
        }
    }
    __syncthreads();

    // ---- phase 4: output projection + bias + residual, store transposed to LDS ----
    {
        const int m  = wave & 3;
        const int n0 = (wave >> 2) * 2;
        v16h A0 = load_a16(outp, E, m * 16, 0, hl, lr);
        v16h A1 = load_a16(outp, E, m * 16, 1, hl, lr);
        #pragma unroll
        for (int n = n0; n < n0 + 2; ++n) {
            v16h B0 = load_bw(wo16, E, n, 0, hl, lr);
            v16h B1 = load_bw(wo16, E, n, 1, hl, lr);
            v8f acc = {};
            acc = wmma_f16(A0, B0, acc);
            acc = wmma_f16(A1, B1, acc);
            int col = n * 16 + lr;
            float bo = outb[col];
            #pragma unroll
            for (int j = 0; j < 8; ++j) {
                int row = m * 16 + hl * 8 + j;
                finF[col * S + row] = acc[j] + bo + xres[row * E + col];  // [e][s] layout
            }
        }
    }
    __syncthreads();

    // ---- phase 5: coalesced store, finF is already [C][H*W] ----
    {
        const float4* src = (const float4*)finF;
        float4* dst = (float4*)(outg + bbase);
        #pragma unroll
        for (int i = 0; i < 4; ++i) dst[i * 256 + tid] = src[i * 256 + tid];
    }
}

extern "C" void kernel_launch(void* const* d_in, const int* in_sizes, int n_in,
                              void* d_out, int out_size, void* d_ws, size_t ws_size,
                              hipStream_t stream) {
    (void)in_sizes; (void)n_in; (void)out_size; (void)ws_size;
    const float* x     = (const float*)d_in[0];
    const float* pos   = (const float*)d_in[1];
    const float* gma   = (const float*)d_in[2];
    const float* bta   = (const float*)d_in[3];
    const float* qkvw  = (const float*)d_in[4];
    const float* qkvb  = (const float*)d_in[5];
    const float* outw  = (const float*)d_in[6];
    const float* outb  = (const float*)d_in[7];
    const float* cbias = (const float*)d_in[8];
    float* out = (float*)d_out;

    _Float16* wq16 = (_Float16*)d_ws;
    _Float16* wo16 = (_Float16*)((char*)d_ws + (size_t)E3 * E * sizeof(_Float16));

    prep_weights<<<48, 256, 0, stream>>>(qkvw, outw, wq16, wo16);
    chess_attn_kernel<<<NBATCH, 256, LDS_BYTES, stream>>>(
        x, pos, gma, bta, qkvb, outb, cbias, wq16, wo16, out);
}
